// TransformerBlock_67078799229426
// MI455X (gfx1250) — matrix operand order repaired and checked by reading, not compiled, not run on hardware
//
#include <hip/hip_runtime.h>
#include <hip/hip_bf16.h>

#define USE_ASYNC_COPY 1   // global_load_async_to_lds_b128 staging (CDNA5)

// ---------------------------------------------------------------------------
// Types for CDNA5 WMMA (wave32)
// ---------------------------------------------------------------------------
typedef __attribute__((ext_vector_type(16))) __bf16         v16bf;
typedef __attribute__((ext_vector_type(16))) unsigned short v16u;
typedef __attribute__((ext_vector_type(8)))  float          v8f;

__device__ __forceinline__ unsigned short f2bf(float f) {
    union { float f; unsigned u; } un; un.f = f;
    unsigned r = un.u + 0x7FFFu + ((un.u >> 16) & 1u);   // round-to-nearest-even
    return (unsigned short)(r >> 16);
}

__device__ __forceinline__ v8f wmma_bf16(v16u a, v16u b, v8f c) {
    return __builtin_amdgcn_wmma_f32_16x16x32_bf16(
        false, __builtin_bit_cast(v16bf, a),
        false, __builtin_bit_cast(v16bf, b),
        (short)0, c, false, false);
}

// 16-byte global -> LDS copy, async on CDNA5 (ASYNCcnt), sync fallback.
__device__ __forceinline__ void copy16_g2l(const void* g, void* lds) {
#if USE_ASYNC_COPY
    unsigned loff = (unsigned)(unsigned long long)lds;   // LDS aperture: addr[31:0]
    asm volatile("global_load_async_to_lds_b128 %0, %1, off"
                 :: "v"(loff), "v"(g) : "memory");
#else
    *reinterpret_cast<uint4*>(lds) = *reinterpret_cast<const uint4*>(g);
#endif
}

// Wait until at most `n` async copies remain in flight (in-order completion).
__device__ __forceinline__ void copy16_wait0() {
#if USE_ASYNC_COPY
    asm volatile("s_wait_asynccnt 0x0" ::: "memory");
#endif
}
__device__ __forceinline__ void copy16_wait4() {
#if USE_ASYNC_COPY
    asm volatile("s_wait_asynccnt 0x4" ::: "memory");
#endif
}

// ---------------------------------------------------------------------------
// Weight preprocessing: fp32 -> bf16, transposed to [N][K] for the GEMM B side
// ---------------------------------------------------------------------------
// in: [K][N] fp32 row-major -> out: [N][K] bf16
__global__ __launch_bounds__(256) void k_cast_t(const float* __restrict__ in,
                                                unsigned short* __restrict__ out,
                                                int K, int N) {
    int i = blockIdx.x * 256 + threadIdx.x;       // over N*K outputs
    int n = i / K, kk = i % K;
    out[i] = f2bf(in[(size_t)kk * N + n]);
}

// Wq/Wk/Wv: [H, C, HD] fp32 -> [N=H*HD][K=C] bf16 (out[n][k] = in[h(n)][k][d(n)])
__global__ __launch_bounds__(256) void k_cast_qkv_t(const float* __restrict__ in,
                                                    unsigned short* __restrict__ out,
                                                    int C, int HD) {
    int i = blockIdx.x * 256 + threadIdx.x;       // over C*C outputs
    int n = i / C, kk = i % C;
    int h = n / HD, d = n % HD;
    out[i] = f2bf(in[((size_t)h * C + kk) * HD + d]);
}

// ---------------------------------------------------------------------------
// LayerNorm: one 256-thread block per row; fp32 in -> bf16 out
// ---------------------------------------------------------------------------
__global__ __launch_bounds__(256) void k_ln(const float* __restrict__ x,
                                            const float* __restrict__ g,
                                            const float* __restrict__ beta,
                                            unsigned short* __restrict__ out, int C) {
    __shared__ float r1[256], r2[256];
    const int row = blockIdx.x;
    const float* xr = x + (size_t)row * C;
    float s = 0.f, s2 = 0.f;
    for (int c = threadIdx.x; c < C; c += 256) { float v = xr[c]; s += v; s2 += v * v; }
    r1[threadIdx.x] = s; r2[threadIdx.x] = s2;
    __syncthreads();
    for (int off = 128; off > 0; off >>= 1) {
        if ((int)threadIdx.x < off) {
            r1[threadIdx.x] += r1[threadIdx.x + off];
            r2[threadIdx.x] += r2[threadIdx.x + off];
        }
        __syncthreads();
    }
    const float mu  = r1[0] / C;
    const float var = r2[0] / C - mu * mu;
    const float rs  = rsqrtf(var + 1e-5f);
    for (int c = threadIdx.x; c < C; c += 256)
        out[(size_t)row * C + c] = f2bf((xr[c] - mu) * rs * g[c] + beta[c]);
}

// ---------------------------------------------------------------------------
// WMMA GEMM, software-pipelined (double-buffered LDS + async in-flight tile):
//   C[M,N] = A[M,K](bf16, row-major) * Bt[N,K](bf16, pre-transposed)
//   (+bias)(+resid fp32)(+gelu); out fp32 or bf16, optionally transposed [N][M].
// Block tile 128x128x32, 256 threads = 8 waves; wave tile 32x64 (2x4 WMMA accs)
// ---------------------------------------------------------------------------
template <bool BIAS, bool RES, bool GELU_, bool OUTBF, bool OUTT>
__global__ __launch_bounds__(256) void k_gemm(const unsigned short* __restrict__ A,
                                              const unsigned short* __restrict__ Bt,
                                              const float* __restrict__ bias,
                                              const float* __restrict__ resid,
                                              void* __restrict__ outp,
                                              int M, int N, int K) {
    __shared__ alignas(16) unsigned short As[2][128 * 32];   // [m][k]
    __shared__ alignas(16) unsigned short Bs[2][128 * 32];   // [n][k]

    const int tid  = threadIdx.x;
    const int wave = tid >> 5;
    const int lane = tid & 31;
    const int wm = wave >> 1, wn = wave & 1;
    const int rowBase = blockIdx.y * 128;
    const int colBase = blockIdx.x * 128;
    const int lg = lane >> 4;      // half-wave group
    const int ll = lane & 15;

    // per-thread staging addresses (4 x 16B copies per tile)
    const int e0 = tid * 8,  r0 = e0 >> 5, c0 = e0 & 31;
    const int e1 = e0 + 2048, r1 = e1 >> 5, c1 = e1 & 31;

    v8f acc[2][4];
#pragma unroll
    for (int mt = 0; mt < 2; ++mt)
#pragma unroll
        for (int nt = 0; nt < 4; ++nt) acc[mt][nt] = (v8f){};

    // prologue: stage tile 0 into buffer 0
    copy16_g2l(&A [(size_t)(rowBase + r0) * K + c0], &As[0][e0]);
    copy16_g2l(&Bt[(size_t)(colBase + r0) * K + c0], &Bs[0][e0]);
    copy16_g2l(&A [(size_t)(rowBase + r1) * K + c1], &As[0][e1]);
    copy16_g2l(&Bt[(size_t)(colBase + r1) * K + c1], &Bs[0][e1]);

    for (int k0 = 0; k0 < K; k0 += 32) {
        const int  buf     = (k0 >> 5) & 1;
        const bool hasNext = (k0 + 32) < K;
        if (hasNext) {                       // stage tile k+1 into other buffer
            const int kn = k0 + 32;
            copy16_g2l(&A [(size_t)(rowBase + r0) * K + kn + c0], &As[buf ^ 1][e0]);
            copy16_g2l(&Bt[(size_t)(colBase + r0) * K + kn + c0], &Bs[buf ^ 1][e0]);
            copy16_g2l(&A [(size_t)(rowBase + r1) * K + kn + c1], &As[buf ^ 1][e1]);
            copy16_g2l(&Bt[(size_t)(colBase + r1) * K + kn + c1], &Bs[buf ^ 1][e1]);
            copy16_wait4();                  // tile k resident; k+1 still in flight
        } else {
            copy16_wait0();
        }
        __syncthreads();

        const unsigned short* Ab = As[buf];
        const unsigned short* Bb = Bs[buf];
        v16u afrag[2];
#pragma unroll
        for (int mt = 0; mt < 2; ++mt) {       // A frag 16x32 (contiguous k = 16*lg + e)
            int am = wm * 32 + mt * 16 + ll;
            int kb = lg * 16;
#pragma unroll
            for (int e = 0; e < 16; ++e)
                afrag[mt][e] = Ab[am * 32 + kb + e];
        }
#pragma unroll
        for (int nt = 0; nt < 4; ++nt) {       // B frag 32x16 (contiguous k = 16*lg + e, as A)
            v16u bfrag;
            int bn = wn * 64 + nt * 16 + ll;
            int kb = lg * 16;
#pragma unroll
            for (int e = 0; e < 16; ++e)
                bfrag[e] = Bb[bn * 32 + kb + e];   // contiguous -> ds_load_b128 x2
#pragma unroll
            for (int mt = 0; mt < 2; ++mt)
                acc[mt][nt] = wmma_bf16(afrag[mt], bfrag, acc[mt][nt]);
        }
        __syncthreads();   // all reads of buf done before it is re-staged
    }

    // Epilogue: C frag layout m = r + 8*lg, n = ll
#pragma unroll
    for (int mt = 0; mt < 2; ++mt)
#pragma unroll
        for (int nt = 0; nt < 4; ++nt)
#pragma unroll
            for (int r = 0; r < 8; ++r) {
                int m = rowBase + wm * 32 + mt * 16 + 8 * lg + r;
                int n = colBase + wn * 64 + nt * 16 + ll;
                float vv = acc[mt][nt][r];
                if (BIAS) vv += bias[n];
                if (RES)  vv += resid[(size_t)m * N + n];
                if (GELU_) vv = 0.5f * vv * (1.f + erff(vv * 0.70710678118654752f));
                if (OUTT)       ((unsigned short*)outp)[(size_t)n * M + m] = f2bf(vv);
                else if (OUTBF) ((unsigned short*)outp)[(size_t)m * N + n] = f2bf(vv);
                else            ((float*)outp)[(size_t)m * N + n] = vv;
            }
}

// ---------------------------------------------------------------------------
// Flash attention (causal): block = (b,h, 128-query tile); 8 waves x 16 rows.
// q/k: bf16 [B*T, C] (head h at cols h*128..); vT: bf16 [C][B*T]; o: [B*T, C]
// ---------------------------------------------------------------------------
__global__ __launch_bounds__(256) void k_attn(const unsigned short* __restrict__ q,
                                              const unsigned short* __restrict__ k,
                                              const unsigned short* __restrict__ vT,
                                              unsigned short* __restrict__ o,
                                              int T, int H, int Mtot) {
    __shared__ alignas(16) unsigned short Qs [128 * 128];  // [q][d]   32 KB
    __shared__ alignas(16) unsigned short Ks [32 * 128];   // [key][d]  8 KB
    __shared__ alignas(16) unsigned short Vts[128 * 32];   // [d][key]  8 KB
    __shared__ alignas(16) unsigned short Ps [128 * 32];   // [q][key]  8 KB

    const int bh = blockIdx.x;
    const int b = bh / H, h = bh % H;
    const int qt = blockIdx.y;
    const int tid = threadIdx.x;
    const int wave = tid >> 5, lane = tid & 31;
    const int lg = lane >> 4, ll = lane & 15;
    const int C = H * 128;
    const int rowg = b * T + qt * 128;
    const int colb = h * 128;

#pragma unroll
    for (int it = 0; it < 8; ++it) {       // Q tile 128x128
        int e = (tid + it * 256) * 8;
        int r = e >> 7, c = e & 127;
        copy16_g2l(&q[(size_t)(rowg + r) * C + colb + c], &Qs[e]);
    }

    float rm[8], rl[8];
    v8f oacc[8];
#pragma unroll
    for (int r = 0; r < 8; ++r) { rm[r] = -__builtin_inff(); rl[r] = 0.f; }
#pragma unroll
    for (int n8 = 0; n8 < 8; ++n8) oacc[n8] = (v8f){};

    const float scale = 0.08838834764831845f;   // 1/sqrt(128)
    const int nkt = (qt + 1) * 4;               // 32-key tiles
    for (int kt = 0; kt < nkt; ++kt) {
        __syncthreads();
        const int krg = b * T + kt * 32;
#pragma unroll
        for (int it = 0; it < 2; ++it) {   // K tile 32x128 ([key][d])
            int e = (tid + it * 256) * 8;
            int r = e >> 7, c = e & 127;
            copy16_g2l(&k[(size_t)(krg + r) * C + colb + c], &Ks[e]);
        }
#pragma unroll
        for (int it = 0; it < 2; ++it) {   // V tile 128x32 ([d][key]) from vT
            int e = (tid + it * 256) * 8;
            int r = e >> 5, c = e & 31;
            copy16_g2l(&vT[(size_t)(colb + r) * Mtot + krg + c], &Vts[e]);
        }
        copy16_wait0();
        __syncthreads();

        // ---- S = Q K^T : 16(q) x 32(keys), K-dim = HD = 128 (4 steps)
        v8f sacc[2];
        sacc[0] = (v8f){}; sacc[1] = (v8f){};
#pragma unroll
        for (int ks = 0; ks < 4; ++ks) {
            v16u af;
            int am = wave * 16 + ll;
            int kb = ks * 32 + lg * 16;
#pragma unroll
            for (int e = 0; e < 16; ++e)
                af[e] = Qs[am * 128 + kb + e];
#pragma unroll
            for (int nt = 0; nt < 2; ++nt) {
                v16u bf_;
                int key = nt * 16 + ll;          // n = key
                int kd  = ks * 32 + lg * 16;     // k-dim = head dim
#pragma unroll
                for (int e = 0; e < 16; ++e)
                    bf_[e] = Ks[key * 128 + kd + e];   // contiguous
                sacc[nt] = wmma_bf16(af, bf_, sacc[nt]);
            }
        }

        // ---- online softmax (fp32), causal mask, P -> LDS (bf16)
#pragma unroll
        for (int r = 0; r < 8; ++r) {
            const int q_abs = qt * 128 + wave * 16 + 8 * lg + r;
            float mloc = -__builtin_inff();
#pragma unroll
            for (int nt = 0; nt < 2; ++nt) {
                int k_abs = kt * 32 + nt * 16 + ll;
                float s = sacc[nt][r] * scale;
                s = (k_abs <= q_abs) ? s : -__builtin_inff();
                sacc[nt][r] = s;
                mloc = fmaxf(mloc, s);
            }
#pragma unroll
            for (int m2 = 1; m2 <= 8; m2 <<= 1)
                mloc = fmaxf(mloc, __shfl_xor(mloc, m2, 32));
            const float mnew  = fmaxf(rm[r], mloc);
            const float alpha = __expf(rm[r] - mnew);
            rm[r] = mnew;
            float psum = 0.f;
#pragma unroll
            for (int nt = 0; nt < 2; ++nt) {
                float p = __expf(sacc[nt][r] - mnew);
                sacc[nt][r] = p;
                psum += p;
            }
#pragma unroll
            for (int m2 = 1; m2 <= 8; m2 <<= 1)
                psum += __shfl_xor(psum, m2, 32);
            rl[r] = rl[r] * alpha + psum;
#pragma unroll
            for (int n8 = 0; n8 < 8; ++n8) oacc[n8][r] *= alpha;
            const int prow = wave * 16 + 8 * lg + r;
#pragma unroll
            for (int nt = 0; nt < 2; ++nt)
                Ps[prow * 32 + nt * 16 + ll] = f2bf(sacc[nt][r]);
        }

        // ---- O += P V : 16(q) x 128(hd), K-dim = 32 keys (1 step)
        {
            v16u af;
            int am = wave * 16 + ll;
            int kb = lg * 16;
#pragma unroll
            for (int e = 0; e < 16; ++e)
                af[e] = Ps[am * 32 + kb + e];
#pragma unroll
            for (int nt2 = 0; nt2 < 8; ++nt2) {
                v16u bf_;
                int dn = nt2 * 16 + ll;
                int kd = lg * 16;
#pragma unroll
                for (int e = 0; e < 16; ++e)
                    bf_[e] = Vts[dn * 32 + kd + e];    // contiguous ([d][key])
                oacc[nt2] = wmma_bf16(af, bf_, oacc[nt2]);
            }
        }
    }

    // ---- normalize + store o (bf16, concat-head layout)
#pragma unroll
    for (int r = 0; r < 8; ++r) {
        const float inv = 1.f / rl[r];
        const int grow = rowg + wave * 16 + 8 * lg + r;
#pragma unroll
        for (int nt2 = 0; nt2 < 8; ++nt2)
            o[(size_t)grow * C + colb + nt2 * 16 + ll] = f2bf(oacc[nt2][r] * inv);
    }
}

// ---------------------------------------------------------------------------
// Host launcher
// ---------------------------------------------------------------------------
extern "C" void kernel_launch(void* const* d_in, const int* in_sizes, int n_in,
                              void* d_out, int out_size, void* d_ws, size_t ws_size,
                              hipStream_t stream) {
    const int B = 4, T = 2048, C = 2048, H = 16, F = 4 * C;  // F = 8192
    const size_t NT = (size_t)B * T;                         // 8192 rows

    const float* x    = (const float*)d_in[0];
    const float* Wq   = (const float*)d_in[1];
    const float* Wk   = (const float*)d_in[2];
    const float* Wv   = (const float*)d_in[3];
    const float* Wp   = (const float*)d_in[4];
    const float* bp   = (const float*)d_in[5];
    const float* W1   = (const float*)d_in[6];
    const float* b1   = (const float*)d_in[7];
    const float* W2   = (const float*)d_in[8];
    const float* b2   = (const float*)d_in[9];
    const float* g1   = (const float*)d_in[10];
    const float* be1  = (const float*)d_in[11];
    const float* g2   = (const float*)d_in[12];
    const float* be2  = (const float*)d_in[13];

    char* w = (char*)d_ws;
    size_t off = 0;
    auto alloc = [&](size_t bytes) -> void* {
        void* p = w + off;
        off = (off + bytes + 255) & ~(size_t)255;
        return p;
    };
    unsigned short* h1  = (unsigned short*)alloc(NT * C * 2);
    unsigned short* qb  = (unsigned short*)alloc(NT * C * 2);
    unsigned short* kb  = (unsigned short*)alloc(NT * C * 2);
    unsigned short* vTb = (unsigned short*)alloc(NT * C * 2);   // [C][B*T]
    unsigned short* ob  = (unsigned short*)alloc(NT * C * 2);
    float*          x1  = (float*)alloc(NT * C * 4);
    unsigned short* h2  = (unsigned short*)alloc(NT * C * 2);
    unsigned short* fb  = (unsigned short*)alloc(NT * F * 2);
    unsigned short* WqT = (unsigned short*)alloc((size_t)C * C * 2);
    unsigned short* WkT = (unsigned short*)alloc((size_t)C * C * 2);
    unsigned short* WvT = (unsigned short*)alloc((size_t)C * C * 2);
    unsigned short* WpT = (unsigned short*)alloc((size_t)C * C * 2);
    unsigned short* W1T = (unsigned short*)alloc((size_t)C * F * 2);
    unsigned short* W2T = (unsigned short*)alloc((size_t)F * C * 2);

    const int CC = C * C;          // 4M
    const int CF = C * F;          // 16M
    k_cast_qkv_t<<<CC / 256, 256, 0, stream>>>(Wq, WqT, C, 128);
    k_cast_qkv_t<<<CC / 256, 256, 0, stream>>>(Wk, WkT, C, 128);
    k_cast_qkv_t<<<CC / 256, 256, 0, stream>>>(Wv, WvT, C, 128);
    k_cast_t<<<CC / 256, 256, 0, stream>>>(Wp, WpT, C, C);   // [C][C] -> [C][C]^T
    k_cast_t<<<CF / 256, 256, 0, stream>>>(W1, W1T, C, F);   // [C][F] -> [F][C]
    k_cast_t<<<CF / 256, 256, 0, stream>>>(W2, W2T, F, C);   // [F][C] -> [C][F]

    // LN1
    k_ln<<<(int)NT, 256, 0, stream>>>(x, g1, be1, h1, C);

    // QKV projections (M=8192, N=2048, K=2048); V written transposed
    dim3 gq(C / 128, (int)NT / 128);
    k_gemm<false, false, false, true , false><<<gq, 256, 0, stream>>>(h1, WqT, nullptr, nullptr, qb,  (int)NT, C, C);
    k_gemm<false, false, false, true , false><<<gq, 256, 0, stream>>>(h1, WkT, nullptr, nullptr, kb,  (int)NT, C, C);
    k_gemm<false, false, false, true , true ><<<gq, 256, 0, stream>>>(h1, WvT, nullptr, nullptr, vTb, (int)NT, C, C);

    // causal flash attention
    k_attn<<<dim3(B * H, T / 128), 256, 0, stream>>>(qb, kb, vTb, ob, T, H, (int)NT);

    // output projection + residual: x1 = x + o @ Wp + bp  (fp32 out)
    k_gemm<true, true, false, false, false><<<gq, 256, 0, stream>>>(ob, WpT, bp, x, x1, (int)NT, C, C);

    // LN2
    k_ln<<<(int)NT, 256, 0, stream>>>(x1, g2, be2, h2, C);

    // FFN1 + GELU (M=8192, N=8192, K=2048) -> bf16
    dim3 gf(F / 128, (int)NT / 128);
    k_gemm<true, false, true, true, false><<<gf, 256, 0, stream>>>(h2, W1T, b1, nullptr, fb, (int)NT, F, C);

    // FFN2 + residual (M=8192, N=2048, K=8192) -> fp32 d_out
    k_gemm<true, true, false, false, false><<<gq, 256, 0, stream>>>(fb, W2T, b2, x1, (float*)d_out, (int)NT, C, F);

    (void)in_sizes; (void)n_in; (void)out_size; (void)ws_size;
}
